// FeatureFusion_10814727651813
// MI455X (gfx1250) — matrix operand order repaired and checked by reading, not compiled, not run on hardware
//
#include <hip/hip_runtime.h>
#include <hip/hip_bf16.h>
#include <cmath>

typedef __attribute__((ext_vector_type(16))) _Float16 v16h;
typedef __attribute__((ext_vector_type(8)))  _Float16 v8h;
typedef __attribute__((ext_vector_type(8)))  float    v8f;

union FragH { v16h v; v8h h8[2]; };

// Problem constants (fixed by the reference)
constexpr int B    = 32;
constexpr int C1   = 256;
constexpr int C2   = 256;
constexpr int C    = 512;    // c1 + c2
constexpr int HW   = 1024;   // 32*32
constexpr int D    = 256;    // projection dim
constexpr int COUT = 256;
constexpr int M_ALL = B * C; // 16384 combined rows for q/k GEMM
constexpr float NEGV = -1e9f;

// ---------------- fragment loaders (wave32 WMMA layouts, ISA 7.12.2) -------

// A 16x32 f16 fragment, row-major source, base points at (m0, k0), ld in elems
__device__ inline v16h load_a16x32(const _Float16* __restrict__ base, int ld, int lane) {
    int r = lane & 15, h = lane >> 4;
    const _Float16* p = base + (size_t)r * ld + h * 8;
    FragH f;
    f.h8[0] = *(const v8h*)(p);        // K = h*8 .. h*8+7
    f.h8[1] = *(const v8h*)(p + 16);   // K = 16 + h*8 .. +7
    return f.v;
}

// B 32x16 f16 fragment where B[k][n] = src[n*ld + k] (src is n-major,
// k-contiguous). base points at (k0, n0) i.e. src + n0*ld + k0.
__device__ inline v16h load_b32x16T(const _Float16* __restrict__ base, int ld, int lane) {
    int cn = lane & 15, h = lane >> 4;
    const _Float16* p = base + (size_t)cn * ld + h * 8;
    FragH f;
    f.h8[0] = *(const v8h*)(p);        // K = h*8 .. h*8+7
    f.h8[1] = *(const v8h*)(p + 16);   // K = 16 + h*8 .. +7
    return f.v;
}

#define WMMA_F16(A, Bf, Acc) \
    __builtin_amdgcn_wmma_f32_16x16x32_f16(false, (A), false, (Bf), (short)0, (Acc), false, false)

// ---------------- kernel 0: f32 -> f16 conversion --------------------------
__global__ void cvt_f16_kernel(const float* __restrict__ src,
                               _Float16* __restrict__ dst, int n) {
    int i = blockIdx.x * blockDim.x + threadIdx.x;
    if (i < n) dst[i] = (_Float16)src[i];
}

// ---------------- kernel 1: bilinear upsample + concat ---------------------
// builds x in both f32 ([b][c][hw]) and f16 forms
__global__ void fuse_kernel(const float* __restrict__ x1, const float* __restrict__ x2,
                            float* __restrict__ xf, _Float16* __restrict__ xh) {
    int i = blockIdx.x * blockDim.x + threadIdx.x;   // over B*C*HW = 2^24
    int p = i & (HW - 1);
    int c = (i >> 10) & (C - 1);
    int b = i >> 19;
    float v;
    if (c < C1) {
        v = x1[((size_t)b * C1 + c) * HW + p];
    } else {
        // half-pixel bilinear 16x16 -> 32x32 (align_corners = false)
        int oy = p >> 5, ox = p & 31;
        float sy = oy * 0.5f - 0.25f;
        float sx = ox * 0.5f - 0.25f;
        int y0 = (int)floorf(sy); float fy = sy - (float)y0;
        int x0 = (int)floorf(sx); float fx = sx - (float)x0;
        int y0c = min(15, max(0, y0)),     x0c = min(15, max(0, x0));
        int y1c = min(15, max(0, y0 + 1)), x1c = min(15, max(0, x0 + 1));
        const float* src = x2 + ((size_t)b * C2 + (c - C1)) * 256;
        float v00 = src[y0c * 16 + x0c], v01 = src[y0c * 16 + x1c];
        float v10 = src[y1c * 16 + x0c], v11 = src[y1c * 16 + x1c];
        v = v00 * (1.f - fy) * (1.f - fx) + v01 * (1.f - fy) * fx
          + v10 * fy * (1.f - fx)         + v11 * fy * fx;
    }
    xf[i] = v;
    xh[i] = (_Float16)v;
}

// ---------------- kernel 2: q/k projection GEMM (32x32 macro-tile) ---------
// q|k = x[16384,1024] @ W^T[1024,256] + bias.  One wave per 32x32 tile:
// 2 A-frags + 2 B-frags -> 4 WMMAs per K-step (2x register reuse).
// N runs 0..511: first 256 -> Wq/bq -> qh, last 256 -> Wk/bk -> kh.
__global__ __launch_bounds__(128) void gemm_qk_kernel(
    const _Float16* __restrict__ xh,  const _Float16* __restrict__ wqh,
    const _Float16* __restrict__ wkh, const float* __restrict__ bq,
    const float* __restrict__ bk,     _Float16* __restrict__ qh,
    _Float16* __restrict__ kh) {
    int lane = threadIdx.x & 31;
    int tile = blockIdx.x * 4 + (threadIdx.x >> 5);   // 8192 macro-tiles
    int tm = tile >> 4, tn = tile & 15;               // 512 x 16
    int m0 = tm * 32, n0 = tn * 32;
    bool isK = (n0 >= D);
    const _Float16* wsrc = isK ? wkh : wqh;
    int nw0 = isK ? n0 - D : n0;
    v8f acc00 = {}, acc01 = {}, acc10 = {}, acc11 = {};
    for (int k0 = 0; k0 < HW; k0 += 32) {
        v16h a0 = load_a16x32 (xh + (size_t)m0 * HW + k0,        HW, lane);
        v16h a1 = load_a16x32 (xh + (size_t)(m0 + 16) * HW + k0, HW, lane);
        v16h b0 = load_b32x16T(wsrc + (size_t)nw0 * HW + k0,        HW, lane);
        v16h b1 = load_b32x16T(wsrc + (size_t)(nw0 + 16) * HW + k0, HW, lane);
        acc00 = WMMA_F16(a0, b0, acc00);
        acc01 = WMMA_F16(a0, b1, acc01);
        acc10 = WMMA_F16(a1, b0, acc10);
        acc11 = WMMA_F16(a1, b1, acc11);
    }
    const float* bias = isK ? bk : bq;
    _Float16* dst = isK ? kh : qh;
    int cn = lane & 15, h = lane >> 4;
    float bv0 = bias[nw0 + cn];
    float bv1 = bias[nw0 + 16 + cn];
    #pragma unroll
    for (int v = 0; v < 8; v++) {
        int m = m0 + v + 8 * h;
        dst[(size_t)m * D + nw0 + cn]        = (_Float16)(acc00[v] + bv0);
        dst[(size_t)m * D + nw0 + 16 + cn]   = (_Float16)(acc01[v] + bv1);
        dst[(size_t)(m + 16) * D + nw0 + cn]      = (_Float16)(acc10[v] + bv0);
        dst[(size_t)(m + 16) * D + nw0 + 16 + cn] = (_Float16)(acc11[v] + bv1);
    }
}

// ---------------- kernel 3: attention GEMM  att = q @ k^T / 16 -------------
// 32x32 macro-tile per wave, per-batch 16x16 macro grid.
__global__ __launch_bounds__(128) void gemm_att_kernel(
    const _Float16* __restrict__ qh, const _Float16* __restrict__ kh,
    float* __restrict__ att) {
    int lane = threadIdx.x & 31;
    int tile = blockIdx.x * 4 + (threadIdx.x >> 5);   // 32 * 256 macro-tiles
    int b = tile >> 8; int rest = tile & 255;
    int tr = rest >> 4, tc = rest & 15;
    int n0 = tr * 32, m0 = tc * 32;
    const _Float16* qb = qh + (size_t)b * C * D;
    const _Float16* kb = kh + (size_t)b * C * D;
    v8f acc00 = {}, acc01 = {}, acc10 = {}, acc11 = {};
    for (int k0 = 0; k0 < D; k0 += 32) {
        v16h a0 = load_a16x32 (qb + (size_t)n0 * D + k0,        D, lane);
        v16h a1 = load_a16x32 (qb + (size_t)(n0 + 16) * D + k0, D, lane);
        v16h b0 = load_b32x16T(kb + (size_t)m0 * D + k0,        D, lane);
        v16h b1 = load_b32x16T(kb + (size_t)(m0 + 16) * D + k0, D, lane);
        acc00 = WMMA_F16(a0, b0, acc00);
        acc01 = WMMA_F16(a0, b1, acc01);
        acc10 = WMMA_F16(a1, b0, acc10);
        acc11 = WMMA_F16(a1, b1, acc11);
    }
    int cn = lane & 15, h = lane >> 4;
    float* ab = att + (size_t)b * C * C;
    #pragma unroll
    for (int v = 0; v < 8; v++) {
        int n = n0 + v + 8 * h;
        ab[(size_t)n * C + m0 + cn]             = acc00[v] * (1.0f / 16.0f);
        ab[(size_t)n * C + m0 + 16 + cn]        = acc01[v] * (1.0f / 16.0f);
        ab[(size_t)(n + 16) * C + m0 + cn]      = acc10[v] * (1.0f / 16.0f);
        ab[(size_t)(n + 16) * C + m0 + 16 + cn] = acc11[v] * (1.0f / 16.0f);
    }
}

// ---------------- kernel 4: per-row top-16 + sigmoid -----------------------
// one wave32 per attention row; diag masked to NEG before selection
__global__ __launch_bounds__(128) void topk_kernel(
    const float* __restrict__ att, float* __restrict__ tsig,
    int* __restrict__ tidx) {
    int lane = threadIdx.x & 31;
    int row = blockIdx.x * 4 + (threadIdx.x >> 5);  // b*512 + n
    int n = row & (C - 1);
    const float* a = att + (size_t)row * C;
    float vals[16];
    #pragma unroll
    for (int i = 0; i < 16; i++) {
        int col = i * 32 + lane;
        float v = a[col];
        vals[i] = (col == n) ? NEGV : v;
    }
    for (int t = 0; t < 16; t++) {
        float best = -3.0e38f; int bi = 0;
        #pragma unroll
        for (int i = 0; i < 16; i++)
            if (vals[i] > best) { best = vals[i]; bi = i; }
        int bcol = bi * 32 + lane;
        for (int off = 16; off > 0; off >>= 1) {
            float ov = __shfl_xor(best, off, 32);
            int   oc = __shfl_xor(bcol, off, 32);
            if (ov > best || (ov == best && oc < bcol)) { best = ov; bcol = oc; }
        }
        if (lane == (bcol & 31)) vals[bcol >> 5] = -3.0e38f;
        if (lane == 0) {
            tsig[(size_t)row * 16 + t] = 1.0f / (1.0f + __expf(-best));
            tidx[(size_t)row * 16 + t] = bcol;
        }
    }
}

// ---------------- kernel 5: sparse outx = sig@x + x, stored transposed -----
// sigmoid(NEG)==0, so each row mixes only 16 top-k rows + its own diagonal.
// Output layout [b][p][c] in f16 so the conv GEMM's B fragments are contiguous.
__global__ void outx_kernel(const float* __restrict__ xf,
                            const float* __restrict__ att,
                            const float* __restrict__ tsig,
                            const int* __restrict__ tidx,
                            _Float16* __restrict__ oxT) {
    int row = blockIdx.x;             // b*512 + n
    int b = row >> 9, n = row & (C - 1);
    __shared__ float s_sig[16];
    __shared__ int   s_idx[16];
    if (threadIdx.x < 16) {
        s_sig[threadIdx.x] = tsig[(size_t)row * 16 + threadIdx.x];
        s_idx[threadIdx.x] = tidx[(size_t)row * 16 + threadIdx.x];
    }
    __syncthreads();
    float dsig  = 1.0f / (1.0f + __expf(-att[(size_t)row * C + n]));
    float cself = 1.0f + dsig;        // identity + restored-diagonal term
    const float* xb = xf + (size_t)b * C * HW;
    for (int p = threadIdx.x; p < HW; p += blockDim.x) {
        float acc = cself * xb[(size_t)n * HW + p];
        #pragma unroll
        for (int t = 0; t < 16; t++)
            acc += s_sig[t] * xb[(size_t)s_idx[t] * HW + p];
        oxT[((size_t)b * HW + p) * C + n] = (_Float16)acc;
    }
}

// ---------------- kernel 6: 1x1 conv GEMM + BN + ReLU ----------------------
// y[b] = conv_w[256,512] @ outx[b][512,1024]; outx is stored transposed so
// the B fragment (fixed p, contiguous c) is two 16B loads per lane.
// 32x32 macro-tile per wave.
__global__ __launch_bounds__(128) void gemm_conv_kernel(
    const _Float16* __restrict__ cwh, const _Float16* __restrict__ oxT,
    const float* __restrict__ gamma,  const float* __restrict__ beta,
    const float* __restrict__ mean,   const float* __restrict__ var,
    float* __restrict__ out) {
    int lane = threadIdx.x & 31;
    int tile = blockIdx.x * 4 + (threadIdx.x >> 5);   // 32 * 8 * 32 macro-tiles
    int b = tile >> 8; int rest = tile & 255;
    int tm = rest >> 5, tn = rest & 31;
    int o0 = tm * 32, p0 = tn * 32;
    const _Float16* xb = oxT + (size_t)b * HW * C;
    v8f acc00 = {}, acc01 = {}, acc10 = {}, acc11 = {};
    for (int k0 = 0; k0 < C; k0 += 32) {
        v16h a0 = load_a16x32 (cwh + (size_t)o0 * C + k0,        C, lane);
        v16h a1 = load_a16x32 (cwh + (size_t)(o0 + 16) * C + k0, C, lane);
        v16h b0 = load_b32x16T(xb + (size_t)p0 * C + k0,        C, lane);
        v16h b1 = load_b32x16T(xb + (size_t)(p0 + 16) * C + k0, C, lane);
        acc00 = WMMA_F16(a0, b0, acc00);
        acc01 = WMMA_F16(a0, b1, acc01);
        acc10 = WMMA_F16(a1, b0, acc10);
        acc11 = WMMA_F16(a1, b1, acc11);
    }
    int cn = lane & 15, h = lane >> 4;
    float* ob = out + (size_t)b * COUT * HW;
    #pragma unroll
    for (int v = 0; v < 8; v++) {
        int o  = o0 + v + 8 * h;
        int o2 = o + 16;
        float inv  = gamma[o]  * rsqrtf(var[o]  + 1e-5f);
        float inv2 = gamma[o2] * rsqrtf(var[o2] + 1e-5f);
        float y00 = (acc00[v] - mean[o])  * inv  + beta[o];
        float y01 = (acc01[v] - mean[o])  * inv  + beta[o];
        float y10 = (acc10[v] - mean[o2]) * inv2 + beta[o2];
        float y11 = (acc11[v] - mean[o2]) * inv2 + beta[o2];
        ob[(size_t)o  * HW + p0 + cn]      = fmaxf(y00, 0.0f);
        ob[(size_t)o  * HW + p0 + 16 + cn] = fmaxf(y01, 0.0f);
        ob[(size_t)o2 * HW + p0 + cn]      = fmaxf(y10, 0.0f);
        ob[(size_t)o2 * HW + p0 + 16 + cn] = fmaxf(y11, 0.0f);
    }
}

// ---------------- host side ------------------------------------------------
extern "C" void kernel_launch(void* const* d_in, const int* in_sizes, int n_in,
                              void* d_out, int out_size, void* d_ws, size_t ws_size,
                              hipStream_t stream) {
    const float* x1 = (const float*)d_in[0];
    const float* x2 = (const float*)d_in[1];
    const float* Wq = (const float*)d_in[2];
    const float* bq = (const float*)d_in[3];
    const float* Wk = (const float*)d_in[4];
    const float* bk = (const float*)d_in[5];
    const float* cw = (const float*)d_in[6];
    const float* g  = (const float*)d_in[7];
    const float* be = (const float*)d_in[8];
    const float* mn = (const float*)d_in[9];
    const float* vr = (const float*)d_in[10];
    float* out = (float*)d_out;

    // carve workspace (256B aligned chunks)
    char* w = (char*)d_ws;
    auto carve = [&](size_t bytes) -> char* {
        char* p = w;
        w += (bytes + 255) & ~(size_t)255;
        return p;
    };
    float*    xf   = (float*)   carve((size_t)M_ALL * HW * 4);   // 64 MB
    _Float16* xh   = (_Float16*)carve((size_t)M_ALL * HW * 2);   // 32 MB
    _Float16* qh   = (_Float16*)carve((size_t)M_ALL * D  * 2);   //  8 MB
    _Float16* kh   = (_Float16*)carve((size_t)M_ALL * D  * 2);   //  8 MB
    float*    att  = (float*)   carve((size_t)M_ALL * C  * 4);   // 32 MB
    _Float16* oxT  = (_Float16*)carve((size_t)B * HW * C * 2);   // 32 MB
    _Float16* wqh  = (_Float16*)carve((size_t)D * HW * 2);
    _Float16* wkh  = (_Float16*)carve((size_t)D * HW * 2);
    _Float16* cwh  = (_Float16*)carve((size_t)COUT * C * 2);
    float*    tsig = (float*)   carve((size_t)M_ALL * 16 * 4);
    int*      tidx = (int*)     carve((size_t)M_ALL * 16 * 4);

    cvt_f16_kernel<<<(D * HW + 255) / 256, 256, 0, stream>>>(Wq, wqh, D * HW);
    cvt_f16_kernel<<<(D * HW + 255) / 256, 256, 0, stream>>>(Wk, wkh, D * HW);
    cvt_f16_kernel<<<(COUT * C + 255) / 256, 256, 0, stream>>>(cw, cwh, COUT * C);

    fuse_kernel<<<(M_ALL * HW) / 256, 256, 0, stream>>>(x1, x2, xf, xh);

    gemm_qk_kernel<<<2048, 128, 0, stream>>>(xh, wqh, wkh, bq, bk, qh, kh);
    gemm_att_kernel<<<2048, 128, 0, stream>>>(qh, kh, att);
    topk_kernel<<<4096, 128, 0, stream>>>(att, tsig, tidx);
    outx_kernel<<<M_ALL, 256, 0, stream>>>(xf, att, tsig, tidx, oxT);
    gemm_conv_kernel<<<2048, 128, 0, stream>>>(cwh, oxT, g, be, mn, vr, out);
}
